// StreamingSTGCN_Model_33792802684974
// MI455X (gfx1250) — matrix operand order repaired and checked by reading, not compiled, not run on hardware
//
#include <hip/hip_runtime.h>
#include <hip/hip_bf16.h>

typedef __attribute__((ext_vector_type(16))) _Float16 v16h;
typedef __attribute__((ext_vector_type(8)))  _Float16 v8h;
typedef __attribute__((ext_vector_type(8)))  float    v8f;

#define T_LEN 4096
#define BATCH 16
#define BT    (T_LEN * BATCH)
#define IN_STR 72   // halfs per row in input LDS tile (96 rows: 32 history + 64)
#define HM_STR 72   // halfs per row in concat-hidden LDS tile

// packed weight-fragment blob layout (halfs)
#define FS_OFF 0
#define FM_OFF 3072            // 3*2*512
#define FL_OFF 8192            // + 5*2*512
#define FO_OFF 26624           // + 9*2*2*512
#define WPACK_N 30720          // + 2*4*512

// ---------------------------------------------------------------------------
// Kernel 1: adapter + GCN projection + hardcoded A_HAT propagation + relu+mean
// grid = BT/64 blocks, 256 threads. Output: f16 h_pool [BT][64]
// ---------------------------------------------------------------------------
__global__ __launch_bounds__(256)
void k_gcn(const float* __restrict__ x, const float* __restrict__ gw,
           _Float16* __restrict__ hp)
{
    __shared__ __attribute__((aligned(16))) float xs[64 * 220];
    __shared__ float gws[18 * 64];

    const int tid = threadIdx.x;
    const size_t bt0 = (size_t)blockIdx.x * 64;

    for (int e = tid; e < 18 * 64; e += 256) gws[e] = gw[e];
    for (int e = tid; e < 64 * 218; e += 256) {
        int r = e / 218;
        int c = e - r * 218;
        xs[r * 220 + c] = x[bt0 * 218 + e];
    }
    __syncthreads();

    const int d  = tid & 63;   // output channel
    const int tg = tid >> 6;   // wave-uniform time group (LDS broadcast reads)

    float g[18];
#pragma unroll
    for (int c = 0; c < 18; ++c) g[c] = gws[c * 64 + d];

    // A_HAT constants (tree graph, degrees: root=6, chain=3, tip=2)
    const float C00 = 1.0f / 6.0f;      // root self
    const float C03 = 0.2357022604f;    // 1/sqrt(18): root <-> finger start
    const float C33 = 1.0f / 3.0f;      // deg3 <-> deg3
    const float C32 = 0.4082482905f;    // 1/sqrt(6): deg3 <-> tip
    const float C22 = 0.5f;             // tip self

    for (int i = 0; i < 16; ++i) {
        int t = tg * 16 + i;
        const float* xr = &xs[t * 220];

        float h0 = 0.f;
#pragma unroll
        for (int c = 0; c < 18; ++c) h0 = fmaf(xr[c], g[c], h0);

        float hb[20];
#pragma unroll
        for (int w = 0; w < 20; ++w) {
            const float* xb = xr + 18 + w * 10;
            float s = 0.f;
#pragma unroll
            for (int c = 0; c < 10; ++c) s = fmaf(xb[c], g[c], s);
            hb[w] = s;
        }

        float fsum = hb[0] + hb[4] + hb[8] + hb[12] + hb[16];
        float pool = fmaxf(C00 * h0 + C03 * fsum, 0.f);
#pragma unroll
        for (int f = 0; f < 5; ++f) {
            float ha = hb[f * 4 + 0], hm = hb[f * 4 + 1];
            float hc = hb[f * 4 + 2], he = hb[f * 4 + 3];
            pool += fmaxf(C03 * h0 + C33 * (ha + hm), 0.f);
            pool += fmaxf(C33 * (ha + hm + hc), 0.f);
            pool += fmaxf(C33 * (hm + hc) + C32 * he, 0.f);
            pool += fmaxf(C32 * hc + C22 * he, 0.f);
        }
        pool *= (1.0f / 21.0f);
        hp[(bt0 + t) * 64 + d] = (_Float16)pool;
    }
}

// ---------------------------------------------------------------------------
// Kernel 1b: pack one MS-TCN layer's weights into wave32 WMMA B-fragment
// layout, once, into a contiguous f16 blob (so every conv block streams it
// coalesced instead of 30k scattered scalar loads).
// B layout: lane<16 -> N=lane,  K = cbase+0..7,16..23 ; lane>=16 -> N=lane-16, K = cbase+8..15,24..31
// ---------------------------------------------------------------------------
__global__ __launch_bounds__(256)
void k_pack(const float* __restrict__ wS, const float* __restrict__ wM,
            const float* __restrict__ wL, const float* __restrict__ wO,
            _Float16* __restrict__ outp)
{
    const int tid = threadIdx.x;
    for (int e = tid; e < 3 * 2 * 512; e += 256) {
        int h = e & 15, ln = (e >> 4) & 31, cc = (e >> 9) & 1, j = e >> 10;
        int n = ln & 15;
        int c = cc * 32 + (h & 7) + ((ln >> 4) << 3) + ((h >> 3) << 4);
        outp[FS_OFF + e] = (_Float16)wS[(j * 64 + c) * 16 + n];
    }
    for (int e = tid; e < 5 * 2 * 512; e += 256) {
        int h = e & 15, ln = (e >> 4) & 31, cc = (e >> 9) & 1, j = e >> 10;
        int n = ln & 15;
        int c = cc * 32 + (h & 7) + ((ln >> 4) << 3) + ((h >> 3) << 4);
        outp[FM_OFF + e] = (_Float16)wM[(j * 64 + c) * 16 + n];
    }
    for (int e = tid; e < 9 * 2 * 2 * 512; e += 256) {
        int h = e & 15, ln = (e >> 4) & 31, nt = (e >> 9) & 1, cc = (e >> 10) & 1, j = e >> 11;
        int n = nt * 16 + (ln & 15);
        int c = cc * 32 + (h & 7) + ((ln >> 4) << 3) + ((h >> 3) << 4);
        outp[FL_OFF + e] = (_Float16)wL[(j * 64 + c) * 32 + n];
    }
    for (int e = tid; e < 2 * 4 * 512; e += 256) {
        int h = e & 15, ln = (e >> 4) & 31, nt = (e >> 9) & 3, cc = (e >> 11) & 1;
        int n = nt * 16 + (ln & 15);
        int c = cc * 32 + (h & 7) + ((ln >> 4) << 3) + ((h >> 3) << 4);
        outp[FO_OFF + e] = (_Float16)wO[c * 64 + n];
    }
}

// ---------------------------------------------------------------------------
// CDNA5 async global->LDS copy (ASYNCcnt path), GVS addressing:
// saddr = 64-bit SGPR base, vaddr = 32-bit per-lane byte offset,
// dst VGPR = LDS byte address.
// ---------------------------------------------------------------------------
__device__ inline void async_copy_b128(unsigned lds_byte, const void* gbase,
                                       unsigned gbyte)
{
    asm volatile("global_load_async_to_lds_b128 %0, %1, %2 offset:0"
                 :: "v"(lds_byte), "v"(gbyte), "s"(gbase)
                 : "memory");
}

__device__ inline void wait_async0()
{
    asm volatile("s_wait_asynccnt 0x0" ::: "memory");
}

// ---------------------------------------------------------------------------
// A-fragment gather per ISA 16-bit A 16x32 layout:
// lane<16 : halves 0..7 = K c0+0..7,  halves 8..15 = K c0+16..23
// lane>=16: halves 0..7 = K c0+8..15, halves 8..15 = K c0+24..31
// ---------------------------------------------------------------------------
__device__ inline v16h load_a(const _Float16* src, int stride, int row, int c0)
{
    const int lane = threadIdx.x & 31;
    const _Float16* p = src + row * stride + c0 + ((lane & 16) ? 8 : 0);
    v8h lo = *(const v8h*)p;
    v8h hi = *(const v8h*)(p + 16);
    v16h r;
#pragma unroll
    for (int i = 0; i < 8; ++i) { r[i] = lo[i]; r[8 + i] = hi[i]; }
    return r;
}

__device__ inline v8f wmma_f16(v16h a, v16h b, v8f c)
{
    return __builtin_amdgcn_wmma_f32_16x16x32_f16(false, a, false, b,
                                                  (short)0, c, false, false);
}

// ---------------------------------------------------------------------------
// Kernel 2: one MS-TCN block (3 dilated causal convs -> relu -> concat -> 1x1)
// grid = BATCH*64 blocks (64 time-tiles of 64 per batch), 256 threads (8 waves)
// in/out: f16 [BT][64]; wpack = pre-packed f16 fragment blob (WPACK_N halfs)
// ---------------------------------------------------------------------------
__global__ __launch_bounds__(256)
void k_mstcn(const _Float16* __restrict__ in, _Float16* __restrict__ out,
             const _Float16* __restrict__ wpack,
             const float* __restrict__ bS, const float* __restrict__ bM,
             const float* __restrict__ bL, const float* __restrict__ bO)
{
    __shared__ __attribute__((aligned(16))) _Float16 xin[96 * IN_STR];  // input + hout reuse
    __shared__ __attribute__((aligned(16))) _Float16 hmid[64 * HM_STR]; // concat hidden
    __shared__ __attribute__((aligned(32))) _Float16 wfrag[WPACK_N];
    __shared__ float lbS[16], lbM[16], lbL[32], lbO[64];

    const int tid  = threadIdx.x;
    const int lane = tid & 31;
    const int wv   = tid >> 5;
    const int l15  = lane & 15;
    const int b    = blockIdx.x >> 6;
    const int tile = blockIdx.x & 63;
    const int t0g  = tile * 64;
    const size_t base = (size_t)b * T_LEN * 64;

    // -------- async copy of packed weight fragments (61440 B, coalesced) ----
    {
        unsigned lds0 = (unsigned)(unsigned long long)&wfrag[0];
        for (int e = tid; e < WPACK_N / 8; e += 256)          // 3840 x 16B
            async_copy_b128(lds0 + (unsigned)e * 16u, wpack, (unsigned)e * 16u);
    }
    // -------- async copy of input tile (32 history rows + 64 rows) ----------
    {
        unsigned lds0 = (unsigned)(unsigned long long)&xin[0];
        const _Float16* gsrc = in + base;
        for (int e = tid; e < 96 * 8; e += 256) {             // 16B chunks
            int r = e >> 3, k = e & 7;
            int t = t0g - 32 + r;
            if (t >= 0)
                async_copy_b128(lds0 + (unsigned)(r * (IN_STR * 2) + k * 16),
                                gsrc, (unsigned)(t * 128 + k * 16));
        }
    }
    // causal zero-pad rows (only first tile of each batch)
    if (tile == 0) {
        for (int e = tid; e < 32 * 32; e += 256) {
            int r = e >> 5, dw = e & 31;
            *(unsigned*)&xin[r * IN_STR + dw * 2] = 0u;
        }
    }
    if (tid < 16) { lbS[tid] = bS[tid]; lbM[tid] = bM[tid]; }
    if (tid < 32) lbL[tid] = bL[tid];
    if (tid < 64) lbO[tid] = bO[tid];

    wait_async0();
    __syncthreads();

    // -------- step 1: three conv branches -> hmid[64][64] (relu) ------------
#pragma unroll
    for (int pass = 0; pass < 2; ++pass) {
        int tix = wv + pass * 8;           // 16 tiles of 16t x 16n
        int st  = tix & 3;                 // time sub-tile
        int br  = tix >> 2;                // 0:k3 1:k5 2/3:k9 halves
        int rbase = st * 16 + l15 + 32;    // LDS row at shift 0
        v8f acc = {};
        float bias;
        int n0;
        if (br == 0) {
#pragma unroll
            for (int j = 0; j < 3; ++j) {
                int rs = rbase - (2 - j);
#pragma unroll
                for (int cc = 0; cc < 2; ++cc) {
                    v16h a = load_a(xin, IN_STR, rs, cc * 32);
                    v16h w = *(const v16h*)&wfrag[FS_OFF + ((j * 2 + cc) * 32 + lane) * 16];
                    acc = wmma_f16(a, w, acc);
                }
            }
            bias = lbS[l15]; n0 = 0;
        } else if (br == 1) {
#pragma unroll
            for (int j = 0; j < 5; ++j) {
                int rs = rbase - (4 - j) * 2;
#pragma unroll
                for (int cc = 0; cc < 2; ++cc) {
                    v16h a = load_a(xin, IN_STR, rs, cc * 32);
                    v16h w = *(const v16h*)&wfrag[FM_OFF + ((j * 2 + cc) * 32 + lane) * 16];
                    acc = wmma_f16(a, w, acc);
                }
            }
            bias = lbM[l15]; n0 = 16;
        } else {
            int nt = br - 2;
#pragma unroll
            for (int j = 0; j < 9; ++j) {
                int rs = rbase - (8 - j) * 4;
#pragma unroll
                for (int cc = 0; cc < 2; ++cc) {
                    v16h a = load_a(xin, IN_STR, rs, cc * 32);
                    v16h w = *(const v16h*)&wfrag[FL_OFF + (((j * 2 + cc) * 2 + nt) * 32 + lane) * 16];
                    acc = wmma_f16(a, w, acc);
                }
            }
            bias = lbL[nt * 16 + l15]; n0 = 32 + nt * 16;
        }
        // C layout: VGPR r -> row M = r + 8*(lane>=16); col N = lane&15
        int rowb = st * 16 + ((lane >> 4) << 3);
        int col  = n0 + l15;
#pragma unroll
        for (int r = 0; r < 8; ++r)
            hmid[(rowb + r) * HM_STR + col] = (_Float16)fmaxf(acc[r] + bias, 0.f);
    }
    __syncthreads();

    // -------- step 2: fused 1x1 conv 64->64, store via xin as staging -------
#pragma unroll
    for (int pass = 0; pass < 2; ++pass) {
        int tix = wv + pass * 8;
        int st  = tix & 3;
        int nt  = tix >> 2;
        int n0  = nt * 16;
        int rb  = st * 16 + l15;
        v8f acc = {};
#pragma unroll
        for (int cc = 0; cc < 2; ++cc) {
            v16h a = load_a(hmid, HM_STR, rb, cc * 32);
            v16h w = *(const v16h*)&wfrag[FO_OFF + ((cc * 4 + nt) * 32 + lane) * 16];
            acc = wmma_f16(a, w, acc);
        }
        float bias = lbO[n0 + l15];
        int rowb = st * 16 + ((lane >> 4) << 3);
        int col  = n0 + l15;
#pragma unroll
        for (int r = 0; r < 8; ++r)
            xin[(rowb + r) * 64 + col] = (_Float16)(acc[r] + bias);
    }
    __syncthreads();

    // -------- coalesced store of 64x64 f16 tile -----------------------------
    for (int e = tid; e < 64 * 32; e += 256) {
        int r = e >> 5, dw = e & 31;
        ((unsigned*)(out + base + (size_t)(t0g + r) * 64))[dw] =
            *(const unsigned*)&xin[r * 64 + dw * 2];
    }
}

// ---------------------------------------------------------------------------
// Kernel 3: out[b][d] = mean_t tanh(t2[b,t,:] . wd[:,d] + bd[d])
// grid = 16 blocks (one per batch), 256 threads
// ---------------------------------------------------------------------------
__global__ __launch_bounds__(256)
void k_head(const _Float16* __restrict__ t2, const float* __restrict__ wd,
            const float* __restrict__ bd, float* __restrict__ outp)
{
    __shared__ float wds[64 * 32];
    __shared__ float bds[32];
    __shared__ float red[256 * 32];

    const int tid = threadIdx.x;
    const int b   = blockIdx.x;

    for (int e = tid; e < 64 * 32; e += 256) wds[e] = wd[e];
    if (tid < 32) bds[tid] = bd[tid];
    __syncthreads();

    float acc[32];
#pragma unroll
    for (int d = 0; d < 32; ++d) acc[d] = 0.f;

    for (int t = tid; t < T_LEN; t += 256) {
        const _Float16* row = t2 + ((size_t)b * T_LEN + t) * 64;
        float xr[64];
#pragma unroll
        for (int c = 0; c < 64; ++c) xr[c] = (float)row[c];
#pragma unroll 4
        for (int d = 0; d < 32; ++d) {
            float s = bds[d];
#pragma unroll
            for (int c = 0; c < 64; ++c) s = fmaf(xr[c], wds[c * 32 + d], s);
            acc[d] += tanhf(s);
        }
    }

#pragma unroll
    for (int d = 0; d < 32; ++d) red[tid * 32 + d] = acc[d];
    __syncthreads();
    for (int s = 128; s > 0; s >>= 1) {
        if (tid < s) {
#pragma unroll
            for (int d = 0; d < 32; ++d)
                red[tid * 32 + d] += red[(tid + s) * 32 + d];
        }
        __syncthreads();
    }
    if (tid < 32) outp[b * 32 + tid] = red[tid] * (1.0f / (float)T_LEN);
}

// ---------------------------------------------------------------------------
extern "C" void kernel_launch(void* const* d_in, const int* in_sizes, int n_in,
                              void* d_out, int out_size, void* d_ws, size_t ws_size,
                              hipStream_t stream)
{
    const float* x     = (const float*)d_in[0];
    const float* gcn_w = (const float*)d_in[1];
    const float* wd    = (const float*)d_in[2];
    const float* bd    = (const float*)d_in[3];
    // t1: ws bs wm bm wl bl wo bo at 4..11 ; t2: at 12..19

    _Float16* hp  = (_Float16*)d_ws;              // h_pool, later reused for t2
    _Float16* t1  = hp + (size_t)BT * 64;
    _Float16* wp1 = t1 + (size_t)BT * 64;
    _Float16* wp2 = wp1 + WPACK_N;

    k_pack<<<1, 256, 0, stream>>>((const float*)d_in[4], (const float*)d_in[6],
                                  (const float*)d_in[8], (const float*)d_in[10], wp1);
    k_pack<<<1, 256, 0, stream>>>((const float*)d_in[12], (const float*)d_in[14],
                                  (const float*)d_in[16], (const float*)d_in[18], wp2);

    k_gcn<<<BT / 64, 256, 0, stream>>>(x, gcn_w, hp);

    k_mstcn<<<BATCH * 64, 256, 0, stream>>>(
        hp, t1, wp1,
        (const float*)d_in[5], (const float*)d_in[7],
        (const float*)d_in[9], (const float*)d_in[11]);

    _Float16* t2 = hp;  // reuse h_pool region
    k_mstcn<<<BATCH * 64, 256, 0, stream>>>(
        t1, t2, wp2,
        (const float*)d_in[13], (const float*)d_in[15],
        (const float*)d_in[17], (const float*)d_in[19]);

    k_head<<<BATCH, 256, 0, stream>>>(t2, wd, bd, (float*)d_out);
}
